// InfiniAttention_87754771792083
// MI455X (gfx1250) — matrix operand order repaired and checked by reading, not compiled, not run on hardware
//
#include <hip/hip_runtime.h>
#include <hip/hip_bf16.h>
#include <stdint.h>

#define B_ 4
#define N_ 2048
#define DIM_ 64
#define H_ 8
#define DH_ 64

typedef __attribute__((ext_vector_type(16))) _Float16 v16h;
typedef __attribute__((ext_vector_type(8)))  float    v8f;

#if __has_builtin(__builtin_amdgcn_global_load_async_to_lds_b128)
#define HAVE_ASYNC_COPY 1
#else
#define HAVE_ASYNC_COPY 0
#endif

#if HAVE_ASYNC_COPY
// Match the builtin's parameter types exactly: pointer to vector_size(16) int,
// in AS1 (global, prints as __device__) and AS3 (LDS, prints as __shared__).
typedef int v4i_vs __attribute__((vector_size(16)));
typedef __attribute__((address_space(1))) v4i_vs* as1_v4i_p;
typedef __attribute__((address_space(3))) v4i_vs* as3_v4i_p;
#endif

// 16-byte global -> LDS copy; async (ASYNCcnt) when the toolchain exposes it.
static __device__ __forceinline__ void copy_b128_to_lds(const void* g, void* l) {
#if HAVE_ASYNC_COPY
  // low 32 bits of a flat shared pointer are the LDS byte offset (ISA 10.2)
  __builtin_amdgcn_global_load_async_to_lds_b128(
      (as1_v4i_p)(uintptr_t)g, (as3_v4i_p)(uint32_t)(uintptr_t)l, 0, 0);
#else
  const uint32_t* gs = (const uint32_t*)g;
  uint32_t* ls = (uint32_t*)l;
#pragma unroll
  for (int i = 0; i < 4; ++i) ls[i] = gs[i];
#endif
}

static __device__ __forceinline__ void async_wait0() {
#if HAVE_ASYNC_COPY
#if __has_builtin(__builtin_amdgcn_s_wait_asynccnt)
  __builtin_amdgcn_s_wait_asynccnt(0);
#else
  asm volatile("s_wait_asynccnt 0" ::: "memory");
#endif
#endif
}

static __device__ __forceinline__ v8f vzero8() {
  v8f z;
#pragma unroll
  for (int i = 0; i < 8; ++i) z[i] = 0.0f;
  return z;
}

static __device__ __forceinline__ v8f wmma_ff16(v16h a, v16h b, v8f c) {
  // D = A(16x32 f16) x B(32x16 f16) + C(16x16 f32)
  return __builtin_amdgcn_wmma_f32_16x16x32_f16(false, a, false, b, (short)0, c,
                                                false, false);
}

static __device__ __forceinline__ float eluf(float x) {
  return x > 0.0f ? x : (__expf(x) - 1.0f);
}
static __device__ __forceinline__ float sigmf(float x) {
  return 1.0f / (1.0f + __expf(-x));
}

// K index for A-fragment slot i (contiguous k = 16*hi + s, matching B fragments)
static __device__ __forceinline__ int a_k0(int i, int hi) {
  return hi * 16 + 2 * i;
}

// ---- A fragments (16x32, row-major M x K source) ----
static __device__ __forceinline__ v16h a_frag_h(const _Float16* p, int ld) {
  int lane = threadIdx.x & 31, m = lane & 15, hi = lane >> 4;
  v16h a;
#pragma unroll
  for (int i = 0; i < 8; ++i) {
    int k0 = a_k0(i, hi);
    a[2 * i]     = p[m * ld + k0];
    a[2 * i + 1] = p[m * ld + k0 + 1];
  }
  return a;
}
static __device__ __forceinline__ v16h a_frag_h_mul(const _Float16* p, int ld, float s) {
  int lane = threadIdx.x & 31, m = lane & 15, hi = lane >> 4;
  v16h a;
#pragma unroll
  for (int i = 0; i < 8; ++i) {
    int k0 = a_k0(i, hi);
    a[2 * i]     = (_Float16)((float)p[m * ld + k0] * s);
    a[2 * i + 1] = (_Float16)((float)p[m * ld + k0 + 1] * s);
  }
  return a;
}
static __device__ __forceinline__ v16h a_frag_h_elu(const _Float16* p, int ld) {
  int lane = threadIdx.x & 31, m = lane & 15, hi = lane >> 4;
  v16h a;
#pragma unroll
  for (int i = 0; i < 8; ++i) {
    int k0 = a_k0(i, hi);
    a[2 * i]     = (_Float16)eluf((float)p[m * ld + k0]);
    a[2 * i + 1] = (_Float16)eluf((float)p[m * ld + k0 + 1]);
  }
  return a;
}
static __device__ __forceinline__ v16h a_frag_f32(const float* p, int ld) {
  int lane = threadIdx.x & 31, m = lane & 15, hi = lane >> 4;
  v16h a;
#pragma unroll
  for (int i = 0; i < 8; ++i) {
    int k0 = a_k0(i, hi);
    a[2 * i]     = (_Float16)p[m * ld + k0];
    a[2 * i + 1] = (_Float16)p[m * ld + k0 + 1];
  }
  return a;
}
// A fragment from a K-major (K x M) source: element(m,k) = p[k*ld + m]
static __device__ __forceinline__ v16h a_frag_h_T(const _Float16* p, int ld) {
  int lane = threadIdx.x & 31, m = lane & 15, hi = lane >> 4;
  v16h a;
#pragma unroll
  for (int i = 0; i < 8; ++i) {
    int k0 = a_k0(i, hi);
    a[2 * i]     = p[k0 * ld + m];
    a[2 * i + 1] = p[(k0 + 1) * ld + m];
  }
  return a;
}

// ---- B fragments (32x16, K x N) ----
static __device__ __forceinline__ v16h b_frag_h(const _Float16* p, int ld) {
  int lane = threadIdx.x & 31, n = lane & 15, hi = lane >> 4;
  v16h b;
#pragma unroll
  for (int i = 0; i < 16; ++i) b[i] = p[(hi * 16 + i) * ld + n];
  return b;
}
// B element(k,n) = p[n*ld + k]  (source is N x K row-major, i.e. transposed)
static __device__ __forceinline__ v16h b_frag_h_T(const _Float16* p, int ld) {
  int lane = threadIdx.x & 31, n = lane & 15, hi = lane >> 4;
  v16h b;
#pragma unroll
  for (int i = 0; i < 16; ++i) b[i] = p[n * ld + hi * 16 + i];
  return b;
}
static __device__ __forceinline__ v16h b_frag_f32(const float* p, int ld) {
  int lane = threadIdx.x & 31, n = lane & 15, hi = lane >> 4;
  v16h b;
#pragma unroll
  for (int i = 0; i < 16; ++i) b[i] = (_Float16)p[(hi * 16 + i) * ld + n];
  return b;
}
static __device__ __forceinline__ v16h b_frag_f32_T(const float* p, int ld) {
  int lane = threadIdx.x & 31, n = lane & 15, hi = lane >> 4;
  v16h b;
#pragma unroll
  for (int i = 0; i < 16; ++i) b[i] = (_Float16)p[n * ld + hi * 16 + i];
  return b;
}

// C/D tile store (row = r + 8*hi, col = lane&15)
static __device__ __forceinline__ void store_c_h(_Float16* p, int ld, v8f c) {
  int lane = threadIdx.x & 31, n = lane & 15, hi = lane >> 4;
#pragma unroll
  for (int r = 0; r < 8; ++r) p[(r + 8 * hi) * ld + n] = (_Float16)c[r];
}

// ------------------------------------------------------------------
// K1: qh/kh/vh = heads(x @ W^T)  -> (B,H,N,DH) f16 in ws
// ------------------------------------------------------------------
__global__ void __launch_bounds__(256)
k_proj(const float* __restrict__ q, const float* __restrict__ k,
       const float* __restrict__ v, const float* __restrict__ Wq,
       const float* __restrict__ Wk, const float* __restrict__ Wv,
       _Float16* __restrict__ qh, _Float16* __restrict__ kh,
       _Float16* __restrict__ vh) {
  const int sel = blockIdx.y;
  const float* x = sel == 0 ? q : (sel == 1 ? k : v);
  const float* W = sel == 0 ? Wq : (sel == 1 ? Wk : Wv);
  _Float16* dst = sel == 0 ? qh : (sel == 1 ? kh : vh);
  const int chunks = N_ / 128;
  const int b = blockIdx.x / chunks, chunk = blockIdx.x % chunks;
  const int w = threadIdx.x >> 5;
  const int row0 = chunk * 128 + w * 16;
  const float* xp = x + ((size_t)b * N_ + row0) * DIM_;
  v16h a0 = a_frag_f32(xp, DIM_);
  v16h a1 = a_frag_f32(xp + 32, DIM_);
  for (int nt = 0; nt < 32; ++nt) {
    v8f c = vzero8();
    c = wmma_ff16(a0, b_frag_f32_T(W + (size_t)(nt * 16) * DIM_, DIM_), c);
    c = wmma_ff16(a1, b_frag_f32_T(W + (size_t)(nt * 16) * DIM_ + 32, DIM_), c);
    const int h = nt >> 2, d0 = (nt & 3) * 16;
    store_c_h(dst + (((size_t)(b * H_ + h) * N_ + row0) * DH_ + d0), DH_, c);
  }
}

// ------------------------------------------------------------------
// K2: v_term = vh - (elu(kh)@mem)/(elu(kh)*mem_norm); new_norm = rowsum(elu(kh))
// ------------------------------------------------------------------
__global__ void __launch_bounds__(256)
k_vterm(const _Float16* __restrict__ kh, const _Float16* __restrict__ vh,
        const float* __restrict__ mem, const float* __restrict__ mem_norm,
        _Float16* __restrict__ vt, float* __restrict__ newnorm) {
  const int chunks = N_ / 128;
  const int bh = blockIdx.x / chunks, chunk = blockIdx.x % chunks;
  const int h = bh % H_;
  const int w = threadIdx.x >> 5, lane = threadIdx.x & 31;
  const int n = lane & 15, hi = lane >> 4;
  const int row0 = chunk * 128 + w * 16;
  const _Float16* kp = kh + ((size_t)bh * N_ + row0) * DH_;
  v16h a0 = a_frag_h_elu(kp, DH_);
  v16h a1 = a_frag_h_elu(kp + 32, DH_);
  const float* mp = mem + (size_t)h * DH_ * DH_;
#pragma unroll
  for (int et = 0; et < 4; ++et) {
    v8f acc = vzero8();
    acc = wmma_ff16(a0, b_frag_f32(mp + et * 16, DH_), acc);
    acc = wmma_ff16(a1, b_frag_f32(mp + 32 * DH_ + et * 16, DH_), acc);
#pragma unroll
    for (int r = 0; r < 8; ++r) {
      const int row = row0 + r + 8 * hi;
      const int e = et * 16 + n;
      const size_t idx = ((size_t)bh * N_ + row) * DH_ + e;
      const float ekv = eluf((float)kh[idx]);
      const float den = ekv * mem_norm[h * N_ + row];
      vt[idx] = (_Float16)((float)vh[idx] - acc[r] / den);
    }
  }
  // new_norm: lane (hi,m) sums cols [hi*32, hi*32+32) of row m, then fold halves
  const int row = row0 + (lane & 15);
  const _Float16* kr = kh + ((size_t)bh * N_ + row) * DH_ + hi * 32;
  float s = 0.0f;
#pragma unroll
  for (int c = 0; c < 32; ++c) s += eluf((float)kr[c]);
  s += __shfl_xor(s, 16, 32);
  if (hi == 0) newnorm[(size_t)bh * N_ + row] = s;
}

// ------------------------------------------------------------------
// K3: per (b,h):  T = Wg^T @ vh (64x64)   and  new_mem = mem + ek^T @ v_term
//     K = 2048 reduction, 32-row chunks staged in LDS (vh/v_term async).
// ------------------------------------------------------------------
__global__ void __launch_bounds__(256)
k_tmem(const _Float16* __restrict__ kh, const _Float16* __restrict__ vh,
       const _Float16* __restrict__ vt, const float* __restrict__ Wg,
       const float* __restrict__ mem, float* __restrict__ Tws,
       float* __restrict__ newmem) {
  const int bh = blockIdx.x;
  const int h = bh % H_;
  const int tid = threadIdx.x, w = tid >> 5, lane = tid & 31;
  const int n = lane & 15, hi = lane >> 4;
  __shared__ _Float16 wg_s[32 * 64];
  __shared__ _Float16 ek_s[32 * 64];
  __shared__ _Float16 vh_s[32 * 64];
  __shared__ _Float16 vt_s[32 * 64];
  const int i0 = w >> 2, j0 = w & 3;
  const int i1 = (w + 8) >> 2, j1 = (w + 8) & 3;
  v8f t0 = vzero8(), t1 = vzero8(), q0 = vzero8(), q1 = vzero8();
  for (int ks = 0; ks < N_ / 32; ++ks) {
    const int m0 = ks * 32;
    const size_t gbase = ((size_t)bh * N_ + m0) * DH_;
    // async direct copies (already f16 in ws)
    copy_b128_to_lds(vh + gbase + tid * 8, &vh_s[tid * 8]);
    copy_b128_to_lds(vt + gbase + tid * 8, &vt_s[tid * 8]);
    // converted / activated tiles go through VGPRs
#pragma unroll
    for (int jj = 0; jj < 8; ++jj) {
      const int idx = tid + 256 * jj;
      const int r = idx >> 6, c = idx & 63;
      wg_s[idx] = (_Float16)Wg[(size_t)(m0 + r) * DIM_ + c];
      ek_s[idx] = (_Float16)eluf((float)kh[gbase + idx]);
    }
    async_wait0();
    __syncthreads();
    v16h aW0 = a_frag_h_T(wg_s + i0 * 16, 64);
    v16h aW1 = a_frag_h_T(wg_s + i1 * 16, 64);
    v16h aE0 = a_frag_h_T(ek_s + i0 * 16, 64);
    v16h aE1 = a_frag_h_T(ek_s + i1 * 16, 64);
    v16h bV0 = b_frag_h(vh_s + j0 * 16, 64);
    v16h bV1 = b_frag_h(vh_s + j1 * 16, 64);
    v16h bT0 = b_frag_h(vt_s + j0 * 16, 64);
    v16h bT1 = b_frag_h(vt_s + j1 * 16, 64);
    t0 = wmma_ff16(aW0, bV0, t0);
    t1 = wmma_ff16(aW1, bV1, t1);
    q0 = wmma_ff16(aE0, bT0, q0);
    q1 = wmma_ff16(aE1, bT1, q1);
    __syncthreads();
  }
  float* Tp = Tws + (size_t)bh * DH_ * DH_;
  const float* memh = mem + (size_t)h * DH_ * DH_;
  float* nm = newmem + (size_t)bh * DH_ * DH_;
#pragma unroll
  for (int r = 0; r < 8; ++r) {
    const int rl = r + 8 * hi;
    const int e0 = (i0 * 16 + rl) * 64 + j0 * 16 + n;
    const int e1 = (i1 * 16 + rl) * 64 + j1 * 16 + n;
    Tp[e0] = t0[r];
    Tp[e1] = t1[r];
    nm[e0] = memh[e0] + q0[r];
    nm[e1] = memh[e1] + q1[r];
  }
}

// ------------------------------------------------------------------
// K4: flash attention (online softmax) + compressive-memory term
//     headout = (1-sig(sl)) * softmax(qk^T/8) @ vh + sig(sg) * mem_q @ T
// ------------------------------------------------------------------
__global__ void __launch_bounds__(256)
k_attn(const _Float16* __restrict__ qh, const _Float16* __restrict__ kh,
       const _Float16* __restrict__ vh, const float* __restrict__ mem,
       const float* __restrict__ mem_norm, const float* __restrict__ Tws,
       const float* __restrict__ s_local, const float* __restrict__ s_long,
       _Float16* __restrict__ ho) {
  const int chunks = N_ / 128;
  const int bh = blockIdx.x / chunks, chunk = blockIdx.x % chunks;
  const int h = bh % H_;
  const int tid = threadIdx.x, w = tid >> 5, lane = tid & 31;
  const int n = lane & 15, hi = lane >> 4;
  const int row0 = chunk * 128 + w * 16;
  __shared__ _Float16 kbuf[32 * 64];
  __shared__ _Float16 vbuf[32 * 64];
  __shared__ _Float16 pbuf[8][16 * 32];
  __shared__ _Float16 mqbuf[8][16 * 64];
  __shared__ _Float16 tbuf[64 * 64];

  const _Float16* qp = qh + ((size_t)bh * N_ + row0) * DH_;
  v16h aq0 = a_frag_h_mul(qp, DH_, 0.125f);   // fold 1/sqrt(DH) into Q
  v16h aq1 = a_frag_h_mul(qp + 32, DH_, 0.125f);
  v16h ae0 = a_frag_h_elu(qp, DH_);
  v16h ae1 = a_frag_h_elu(qp + 32, DH_);

  float mi[8], li[8];
#pragma unroll
  for (int r = 0; r < 8; ++r) { mi[r] = -1e30f; li[r] = 0.0f; }
  v8f o0 = vzero8(), o1 = vzero8(), o2 = vzero8(), o3 = vzero8();

  const _Float16* kbase = kh + (size_t)bh * N_ * DH_;
  const _Float16* vbase = vh + (size_t)bh * N_ * DH_;
  for (int kc = 0; kc < N_ / 32; ++kc) {
    const _Float16* ksrc = kbase + (size_t)kc * 32 * DH_;
    const _Float16* vsrc = vbase + (size_t)kc * 32 * DH_;
    // async stage K/V chunk: 256 threads x 16B = 2 x (32x64) f16 tiles
    copy_b128_to_lds(ksrc + tid * 8, &kbuf[tid * 8]);
    copy_b128_to_lds(vsrc + tid * 8, &vbuf[tid * 8]);
    if (kc + 1 < N_ / 32) {  // global_prefetch_b8 of next K/V chunk
      __builtin_prefetch(ksrc + 2048 + tid * 8, 0, 1);
      __builtin_prefetch(vsrc + 2048 + tid * 8, 0, 1);
    }
    async_wait0();
    __syncthreads();

    // S = (Q*scale) @ K^T : two 16x16 tiles over the 32 keys
    v8f s0 = vzero8(), s1 = vzero8();
    s0 = wmma_ff16(aq0, b_frag_h_T(kbuf, 64), s0);
    s0 = wmma_ff16(aq1, b_frag_h_T(kbuf + 32, 64), s0);
    s1 = wmma_ff16(aq0, b_frag_h_T(kbuf + 16 * 64, 64), s1);
    s1 = wmma_ff16(aq1, b_frag_h_T(kbuf + 16 * 64 + 32, 64), s1);

    // online softmax across the 16-lane halves
    float alpha[8];
#pragma unroll
    for (int r = 0; r < 8; ++r) {
      float t = fmaxf(s0[r], s1[r]);
#pragma unroll
      for (int msk = 1; msk < 16; msk <<= 1) t = fmaxf(t, __shfl_xor(t, msk, 32));
      const float mn = fmaxf(mi[r], t);
      alpha[r] = __expf(mi[r] - mn);
      const float p0 = __expf(s0[r] - mn);
      const float p1 = __expf(s1[r] - mn);
      float rs = p0 + p1;
#pragma unroll
      for (int msk = 1; msk < 16; msk <<= 1) rs += __shfl_xor(rs, msk, 32);
      li[r] = li[r] * alpha[r] + rs;
      mi[r] = mn;
      s0[r] = p0;
      s1[r] = p1;
    }
#pragma unroll
    for (int r = 0; r < 8; ++r) {
      o0[r] *= alpha[r]; o1[r] *= alpha[r]; o2[r] *= alpha[r]; o3[r] *= alpha[r];
    }
    // C-layout -> A-layout round trip through LDS for P
    _Float16* pw = pbuf[w];
#pragma unroll
    for (int r = 0; r < 8; ++r) {
      const int rl = r + 8 * hi;
      pw[rl * 32 + n] = (_Float16)s0[r];
      pw[rl * 32 + 16 + n] = (_Float16)s1[r];
    }
    v16h ap = a_frag_h(pw, 32);
    o0 = wmma_ff16(ap, b_frag_h(vbuf + 0, 64), o0);
    o1 = wmma_ff16(ap, b_frag_h(vbuf + 16, 64), o1);
    o2 = wmma_ff16(ap, b_frag_h(vbuf + 32, 64), o2);
    o3 = wmma_ff16(ap, b_frag_h(vbuf + 48, 64), o3);
    __syncthreads();
  }

  // ---- compressive memory term: mem_q = (elu(q)@mem)/(elu(q)*norm); out2 = mem_q @ T
  const float* Tp = Tws + (size_t)bh * DH_ * DH_;
#pragma unroll
  for (int jj = 0; jj < 16; ++jj) {
    const int idx = tid + 256 * jj;
    tbuf[idx] = (_Float16)Tp[idx];
  }
  __syncthreads();
  const float* mp = mem + (size_t)h * DH_ * DH_;
  v8f na[4];
#pragma unroll
  for (int et = 0; et < 4; ++et) {
    na[et] = vzero8();
    na[et] = wmma_ff16(ae0, b_frag_f32(mp + et * 16, DH_), na[et]);
    na[et] = wmma_ff16(ae1, b_frag_f32(mp + 32 * DH_ + et * 16, DH_), na[et]);
  }
  _Float16* mq = mqbuf[w];
#pragma unroll
  for (int et = 0; et < 4; ++et) {
#pragma unroll
    for (int r = 0; r < 8; ++r) {
      const int rl = r + 8 * hi, row = row0 + rl, e = et * 16 + n;
      const float eqv = eluf((float)qh[((size_t)bh * N_ + row) * DH_ + e]);
      const float den = eqv * mem_norm[h * N_ + row];
      mq[rl * 64 + e] = (_Float16)(na[et][r] / den);
    }
  }
  v8f g0 = vzero8(), g1 = vzero8(), g2 = vzero8(), g3 = vzero8();
#pragma unroll
  for (int ks = 0; ks < 2; ++ks) {
    v16h am = a_frag_h(mq + ks * 32, 64);
    const _Float16* tb = tbuf + ks * 32 * 64;
    g0 = wmma_ff16(am, b_frag_h(tb + 0, 64), g0);
    g1 = wmma_ff16(am, b_frag_h(tb + 16, 64), g1);
    g2 = wmma_ff16(am, b_frag_h(tb + 32, 64), g2);
    g3 = wmma_ff16(am, b_frag_h(tb + 48, 64), g3);
  }
  const float c1 = 1.0f - sigmf(s_local[0]);
  const float c2 = sigmf(s_long[0]);
  _Float16* hop = ho + ((size_t)bh * N_ + row0) * DH_;
#pragma unroll
  for (int r = 0; r < 8; ++r) {
    const int rl = r + 8 * hi;
    const float inv = c1 / li[r];
    hop[rl * 64 + 0 + n]  = (_Float16)(o0[r] * inv + c2 * g0[r]);
    hop[rl * 64 + 16 + n] = (_Float16)(o1[r] * inv + c2 * g1[r]);
    hop[rl * 64 + 32 + n] = (_Float16)(o2[r] * inv + c2 * g2[r]);
    hop[rl * 64 + 48 + n] = (_Float16)(o3[r] * inv + c2 * g3[r]);
  }
}

// ------------------------------------------------------------------
// K5: out = headout @ Wo^T + bo
// ------------------------------------------------------------------
__global__ void __launch_bounds__(256)
k_outproj(const _Float16* __restrict__ ho, const float* __restrict__ Wo,
          const float* __restrict__ bo, float* __restrict__ out) {
  const int chunks = N_ / 128;
  const int b = blockIdx.x / chunks, chunk = blockIdx.x % chunks;
  const int tid = threadIdx.x, w = tid >> 5, lane = tid & 31;
  const int n = lane & 15, hi = lane >> 4;
  const int row0 = chunk * 128 + w * 16;
  v8f c0 = vzero8(), c1 = vzero8(), c2 = vzero8(), c3 = vzero8();
  for (int ks = 0; ks < 16; ++ks) {
    const int cc = ks * 32;
    const int hh = cc >> 6, d0 = cc & 63;
    const _Float16* ap = ho + (((size_t)(b * H_ + hh) * N_ + row0) * DH_ + d0);
    v16h a = a_frag_h(ap, DH_);
    c0 = wmma_ff16(a, b_frag_f32_T(Wo + 0 * 512 + cc, 512), c0);
    c1 = wmma_ff16(a, b_frag_f32_T(Wo + 16 * 512 + cc, 512), c1);
    c2 = wmma_ff16(a, b_frag_f32_T(Wo + 32 * 512 + cc, 512), c2);
    c3 = wmma_ff16(a, b_frag_f32_T(Wo + 48 * 512 + cc, 512), c3);
  }
  float* op = out + ((size_t)b * N_ + row0) * DIM_;
#pragma unroll
  for (int r = 0; r < 8; ++r) {
    const int rl = r + 8 * hi;
    op[rl * 64 + 0 + n]  = c0[r] + bo[0 + n];
    op[rl * 64 + 16 + n] = c1[r] + bo[16 + n];
    op[rl * 64 + 32 + n] = c2[r] + bo[32 + n];
    op[rl * 64 + 48 + n] = c3[r] + bo[48 + n];
  }
}

extern "C" void kernel_launch(void* const* d_in, const int* in_sizes, int n_in,
                              void* d_out, int out_size, void* d_ws, size_t ws_size,
                              hipStream_t stream) {
  (void)in_sizes; (void)n_in; (void)out_size; (void)ws_size;
  const float* q        = (const float*)d_in[0];
  const float* k        = (const float*)d_in[1];
  const float* v        = (const float*)d_in[2];
  const float* Wq       = (const float*)d_in[3];
  const float* Wk       = (const float*)d_in[4];
  const float* Wv       = (const float*)d_in[5];
  const float* Wo       = (const float*)d_in[6];
  const float* bo       = (const float*)d_in[7];
  const float* Wg       = (const float*)d_in[8];
  const float* s_local  = (const float*)d_in[9];
  const float* s_long   = (const float*)d_in[10];
  const float* mem      = (const float*)d_in[11];
  const float* mem_norm = (const float*)d_in[12];

  float* out     = (float*)d_out;                          // (B,N,DIM)
  float* newmem  = out + (size_t)B_ * N_ * DIM_;           // (B,H,DH,DH)
  float* newnorm = newmem + (size_t)B_ * H_ * DH_ * DH_;   // (B,H,N)

  const size_t SZ = (size_t)B_ * H_ * N_ * DH_;
  _Float16* qh = (_Float16*)d_ws;
  _Float16* kh = qh + SZ;
  _Float16* vh = kh + SZ;
  _Float16* vt = vh + SZ;
  _Float16* ho = vt + SZ;
  float* Tws = (float*)(ho + SZ);  // (B,H,64,64) f32

  const int chunks = N_ / 128;
  k_proj<<<dim3(B_ * chunks, 3), 256, 0, stream>>>(q, k, v, Wq, Wk, Wv, qh, kh, vh);
  k_vterm<<<B_ * H_ * chunks, 256, 0, stream>>>(kh, vh, mem, mem_norm, vt, newnorm);
  k_tmem<<<B_ * H_, 256, 0, stream>>>(kh, vh, vt, Wg, mem, Tws, newmem);
  k_attn<<<B_ * H_ * chunks, 256, 0, stream>>>(qh, kh, vh, mem, mem_norm, Tws,
                                               s_local, s_long, ho);
  k_outproj<<<B_ * chunks, 256, 0, stream>>>(ho, Wo, bo, out);
}
